// GPT_7060926234971
// MI455X (gfx1250) — compile-verified, matrix operand order checked by reading, not run on hardware
//
#include <hip/hip_runtime.h>
#include <stdint.h>

typedef __attribute__((ext_vector_type(16))) _Float16 v16h;
typedef __attribute__((ext_vector_type(8)))  _Float16 v8h;
typedef __attribute__((ext_vector_type(8)))  float    v8f;
typedef __attribute__((ext_vector_type(4)))  int      v4i;

#define NB      2
#define SEQ     1024
#define ROWS    2048            // NB*SEQ
#define DM      768
#define D3      2304
#define D4      3072
#define NHEAD   12
#define HD      64
#define VOCAB   50257
#define NLAYER  6

#define GEMM_OUT_F16       0
#define GEMM_OUT_F16_GELU  1
#define GEMM_OUT_F32_RESID 2
#define GEMM_OUT_F32       3
#define GEMM_OUT_QKV       4

#define WMMA_F16(Afr, Bfr, Cfr) \
  __builtin_amdgcn_wmma_f32_16x16x32_f16(false, (Afr), false, (Bfr), (short)0, (Cfr), false, false)

// CDNA5 async global->LDS copy path (ASYNCcnt), if the toolchain exposes it.
#if defined(__has_builtin)
#if __has_builtin(__builtin_amdgcn_global_load_async_to_lds_b128) && \
    __has_builtin(__builtin_amdgcn_s_wait_asynccnt)
#define HAVE_ASYNC_LDS 1
#endif
#endif
#ifndef HAVE_ASYNC_LDS
#define HAVE_ASYNC_LDS 0
#endif

#if HAVE_ASYNC_LDS
__device__ __forceinline__ void async_copy_b128(const void* g, void* l) {
  __builtin_amdgcn_global_load_async_to_lds_b128(
      (__attribute__((address_space(1))) v4i*)(g),
      (__attribute__((address_space(3))) v4i*)(l), 0, 0);
}
#endif

// 16-bit fragment: lane's 16 halves = two contiguous 8-half chunks along K:
//   K = hi*8 + {0..7}  and  K = 16 + hi*8 + {0..7}
// -> every fragment load is two 16B vector loads.
#define LOAD_FRAG(dstv, rowptr, hiv)                                           \
  do {                                                                         \
    v8h _lo = *(const v8h*)&(rowptr)[(hiv) * 8];                               \
    v8h _hi = *(const v8h*)&(rowptr)[16 + (hiv) * 8];                          \
    dstv = __builtin_shufflevector(_lo, _hi, 0, 1, 2, 3, 4, 5, 6, 7, 8, 9, 10, \
                                   11, 12, 13, 14, 15);                        \
  } while (0)

// ---------------------------------------------------------------- cvt f32->f16
__global__ void cvt_f32_f16(const float* __restrict__ src,
                            _Float16* __restrict__ dst, long n) {
  long i = (long)blockIdx.x * blockDim.x + threadIdx.x;
  long stride = (long)gridDim.x * blockDim.x;
  for (; i < n; i += stride) dst[i] = (_Float16)src[i];
}

// f32 [L,K,N] -> f16 [L,N,K]  (N-major weights: B staging is K-contiguous)
__global__ void cvt_transpose_f16(const float* __restrict__ src,
                                  _Float16* __restrict__ dst,
                                  int L, int K, int N) {
  long total = (long)L * K * N;
  long i = (long)blockIdx.x * blockDim.x + threadIdx.x;
  long stride = (long)gridDim.x * blockDim.x;
  for (; i < total; i += stride) {
    long l = i / ((long)K * N);
    long rem = i - l * (long)K * N;
    int k = (int)(rem / N);
    int n = (int)(rem - (long)k * N);
    dst[l * K * N + (long)n * K + k] = (_Float16)src[i];
  }
}

// ---------------------------------------------------------------- embedding
__global__ void embed_kernel(const int* __restrict__ idx,
                             const float* __restrict__ wte,
                             const float* __restrict__ wpe,
                             float* __restrict__ x) {
  int i = blockIdx.x * blockDim.x + threadIdx.x;
  if (i >= ROWS * DM) return;
  int row = i / DM, d = i - row * DM;
  int t = row & (SEQ - 1);
  x[i] = wte[(long)idx[row] * DM + d] + wpe[t * DM + d];
}

// ---------------------------------------------------------------- layernorm -> f16
__global__ __launch_bounds__(32)
void layernorm_f16(const float* __restrict__ x,
                   const float* __restrict__ s,
                   const float* __restrict__ b,
                   _Float16* __restrict__ out) {
  int row = blockIdx.x;
  int lane = threadIdx.x;
  const float* xr = x + (long)row * DM;
  float v[24];
  float sum = 0.f, sq = 0.f;
#pragma unroll
  for (int i = 0; i < 24; ++i) {
    float t = xr[lane + 32 * i];
    v[i] = t; sum += t; sq += t * t;
  }
#pragma unroll
  for (int m = 16; m >= 1; m >>= 1) {
    sum += __shfl_xor(sum, m, 32);
    sq  += __shfl_xor(sq,  m, 32);
  }
  float mean = sum * (1.0f / DM);
  float var  = sq * (1.0f / DM) - mean * mean;
  float rs   = rsqrtf(var + 1e-5f);
  _Float16* orow = out + (long)row * DM;
#pragma unroll
  for (int i = 0; i < 24; ++i) {
    int c = lane + 32 * i;
    orow[c] = (_Float16)((v[i] - mean) * rs * s[c] + b[c]);
  }
}

// ---------------------------------------------------------------- WMMA GEMM
// C[M,N] = A[M,K] (f16 row-major) x B (f16, N-major: N x K) + bias, fused
// epilogue. 128 threads = 4 waves; tile 64(M) x 128(N) x 32(K); double-buffered
// LDS with async global->LDS copies overlapping the WMMAs.
__global__ __launch_bounds__(128)
void gemm_wmma(const _Float16* __restrict__ A, const _Float16* __restrict__ B,
               const float* __restrict__ bias, const float* __restrict__ resid,
               float* __restrict__ of32, _Float16* __restrict__ of16,
               _Float16* __restrict__ vt,
               int M, int N, int K, int mode) {
  __shared__ _Float16 As[2][64][40];    // stride 80B -> 16B-aligned chunks
  __shared__ _Float16 BsT[2][128][40];  // [n][k]
  int tid  = threadIdx.x;
  int lane = tid & 31;
  int w    = tid >> 5;
  int lr   = lane & 15;
  int hi   = lane >> 4;
  int n0   = blockIdx.x * 128;
  int m0   = blockIdx.y * 64;
  int fullN = (n0 + 128 <= N);

  v8f acc[8];
#pragma unroll
  for (int j = 0; j < 8; ++j) acc[j] = (v8f){};

  int nkt = K >> 5;

#if HAVE_ASYNC_LDS
#define STAGE_TILES(KT, BUF)                                                   \
  do {                                                                         \
    _Pragma("unroll") for (int i = 0; i < 2; ++i) {                            \
      int u = i * 128 + tid; int r = u >> 2; int c8 = (u & 3) << 3;            \
      async_copy_b128(A + (long)(m0 + r) * K + (KT) * 32 + c8,                 \
                      &As[BUF][r][c8]);                                        \
    }                                                                          \
    if (fullN) {                                                               \
      _Pragma("unroll") for (int i = 0; i < 4; ++i) {                          \
        int u = i * 128 + tid; int n = u >> 2; int c8 = (u & 3) << 3;          \
        async_copy_b128(B + (long)(n0 + n) * K + (KT) * 32 + c8,               \
                        &BsT[BUF][n][c8]);                                     \
      }                                                                        \
    } else {                                                                   \
      _Pragma("unroll") for (int i = 0; i < 4; ++i) {                          \
        int u = i * 128 + tid; int n = u >> 2; int c8 = (u & 3) << 3;          \
        if (n0 + n < N)                                                        \
          async_copy_b128(B + (long)(n0 + n) * K + (KT) * 32 + c8,             \
                          &BsT[BUF][n][c8]);                                   \
      }                                                                        \
    }                                                                          \
  } while (0)

  STAGE_TILES(0, 0);
  for (int kt = 0; kt < nkt; ++kt) {
    int buf = kt & 1;
    __builtin_amdgcn_s_wait_asynccnt(0);  // tile kt resident in LDS
    __syncthreads();                      // visible to all waves; prev compute done
    if (kt + 1 < nkt) STAGE_TILES(kt + 1, buf ^ 1);  // overlap with WMMAs below
    // preload all fragments into distinct registers so the scheduler can
    // pipeline the 18 ds_load_b128 against the 8 back-to-back WMMAs
    v16h a;
    LOAD_FRAG(a, As[buf][w * 16 + lr], hi);
    v16h bf[8];
#pragma unroll
    for (int j = 0; j < 8; ++j) LOAD_FRAG(bf[j], BsT[buf][j * 16 + lr], hi);
#pragma unroll
    for (int j = 0; j < 8; ++j) acc[j] = WMMA_F16(a, bf[j], acc[j]);
  }
#undef STAGE_TILES
#else
  for (int kt = 0; kt < nkt; ++kt) {
#pragma unroll
    for (int i = 0; i < 2; ++i) {
      int u = i * 128 + tid; int r = u >> 2; int c8 = (u & 3) << 3;
      *(v8h*)&As[0][r][c8] = *(const v8h*)(A + (long)(m0 + r) * K + kt * 32 + c8);
    }
#pragma unroll
    for (int i = 0; i < 4; ++i) {
      int u = i * 128 + tid; int n = u >> 2; int c8 = (u & 3) << 3;
      int gn = n0 + n;
      v8h val = (v8h){};
      if (gn < N) val = *(const v8h*)(B + (long)gn * K + kt * 32 + c8);
      *(v8h*)&BsT[0][n][c8] = val;
    }
    __syncthreads();
    v16h a;
    LOAD_FRAG(a, As[0][w * 16 + lr], hi);
    v16h bf[8];
#pragma unroll
    for (int j = 0; j < 8; ++j) LOAD_FRAG(bf[j], BsT[0][j * 16 + lr], hi);
#pragma unroll
    for (int j = 0; j < 8; ++j) acc[j] = WMMA_F16(a, bf[j], acc[j]);
    __syncthreads();
  }
#endif

  // ---- epilogue (C layout: lane col = lr, VGPR r -> row r + 8*hi)
  int rowb = m0 + w * 16 + (hi ? 8 : 0);
#pragma unroll
  for (int j = 0; j < 8; ++j) {
    int col = n0 + j * 16 + lr;
    if (col >= N) continue;
    float bv = bias ? bias[col] : 0.0f;
#pragma unroll
    for (int r = 0; r < 8; ++r) {
      int row = rowb + r;
      long off = (long)row * N + col;
      float v = acc[j][r] + bv;
      if (mode == GEMM_OUT_F16) {
        of16[off] = (_Float16)v;
      } else if (mode == GEMM_OUT_F16_GELU) {
        float c = 0.7978845608028654f * (v + 0.044715f * v * v * v);
        of16[off] = (_Float16)(0.5f * v * (1.0f + tanhf(c)));
      } else if (mode == GEMM_OUT_F32_RESID) {
        of32[off] = resid[off] + v;
      } else if (mode == GEMM_OUT_F32) {
        of32[off] = v;
      } else { // GEMM_OUT_QKV: q,k stay [row][D3]; v scattered to vT[b,h,d][t]
        if (col < 2 * DM) {
          of16[off] = (_Float16)v;
        } else {
          int dg = col - 2 * DM;
          int h  = dg >> 6;
          int dd = dg & 63;
          int bb = row >> 10;
          int t  = row & (SEQ - 1);
          vt[(((long)bb * NHEAD + h) * HD + dd) * SEQ + t] = (_Float16)v;
        }
      }
    }
  }
}

// ---------------------------------------------------------------- flash attention
// 1 wave per 16-query tile per (batch, head).
// q,k in qkv[row][D3] (+0 / +DM); v in vT[((b*NHEAD+h)*HD+d)*SEQ + t].
__global__ __launch_bounds__(32)
void flash_attn(const _Float16* __restrict__ qkv,
                const _Float16* __restrict__ vt,
                _Float16* __restrict__ y) {
  __shared__ _Float16 Pst[16][40];
  int lane = threadIdx.x;
  int lr = lane & 15, hi = lane >> 4;
  int tile = blockIdx.x;
  int qt = tile & (SEQ / 16 - 1);
  int h  = (tile / (SEQ / 16)) % NHEAD;
  int b  = tile / ((SEQ / 16) * NHEAD);
  int qbase = qt * 16;
  long rowbase = (long)b * SEQ;
  const _Float16* vbase = vt + ((long)b * NHEAD + h) * HD * SEQ;
  const float sc = 0.125f;   // 1/sqrt(64)

  // preload Q A-fragments (d 0..31 and 32..63) -- two 16B loads each
  v16h aq0, aq1;
  {
    const _Float16* qp = qkv + (rowbase + qbase + lr) * D3 + h * HD;
    LOAD_FRAG(aq0, qp, hi);
    LOAD_FRAG(aq1, qp + 32, hi);
  }

  float mrow[8], lrow[8];
#pragma unroll
  for (int r = 0; r < 8; ++r) { mrow[r] = -3.0e38f; lrow[r] = 0.0f; }
  v8f o0 = {}, o1 = {}, o2 = {}, o3 = {};

  int kbmax = (qbase + 15) >> 5;
  for (int kb = 0; kb <= kbmax; ++kb) {
    v8f s0 = {}, s1 = {};
    // ---- S = Q @ K^T : d is contiguous per key row -> vector frag loads
    {
      const _Float16* kp0 = qkv + (rowbase + kb * 32 + lr) * D3 + DM + h * HD;
      const _Float16* kp1 = qkv + (rowbase + kb * 32 + 16 + lr) * D3 + DM + h * HD;
      v16h bk00, bk01, bk10, bk11;
      LOAD_FRAG(bk00, kp0, hi);
      LOAD_FRAG(bk01, kp0 + 32, hi);
      LOAD_FRAG(bk10, kp1, hi);
      LOAD_FRAG(bk11, kp1 + 32, hi);
      s0 = WMMA_F16(aq0, bk00, s0);
      s0 = WMMA_F16(aq1, bk01, s0);
      s1 = WMMA_F16(aq0, bk10, s1);
      s1 = WMMA_F16(aq1, bk11, s1);
    }

    // ---- causal mask + online softmax (row stats per 16-lane half)
    float p0[8], p1[8], sfr[8];
#pragma unroll
    for (int r = 0; r < 8; ++r) {
      int q  = qbase + r + (hi ? 8 : 0);
      int k0 = kb * 32 + lr;
      float v0 = (k0      <= q) ? s0[r] * sc : -3.0e38f;
      float v1 = (k0 + 16 <= q) ? s1[r] * sc : -3.0e38f;
      float mx = fmaxf(v0, v1);
#pragma unroll
      for (int m = 8; m >= 1; m >>= 1) mx = fmaxf(mx, __shfl_xor(mx, m, 32));
      float mnew = fmaxf(mrow[r], mx);
      float sf = __expf(mrow[r] - mnew);
      float e0 = __expf(v0 - mnew);
      float e1 = __expf(v1 - mnew);
      float rs = e0 + e1;
#pragma unroll
      for (int m = 8; m >= 1; m >>= 1) rs += __shfl_xor(rs, m, 32);
      lrow[r] = lrow[r] * sf + rs;
      mrow[r] = mnew;
      p0[r] = e0; p1[r] = e1; sfr[r] = sf;
    }
    // rescale O, stage P (C-layout -> LDS -> A-layout)
#pragma unroll
    for (int r = 0; r < 8; ++r) {
      o0[r] *= sfr[r]; o1[r] *= sfr[r]; o2[r] *= sfr[r]; o3[r] *= sfr[r];
      Pst[r + (hi ? 8 : 0)][lr]      = (_Float16)p0[r];
      Pst[r + (hi ? 8 : 0)][16 + lr] = (_Float16)p1[r];
    }
    __syncthreads();
    v16h pa;
    LOAD_FRAG(pa, Pst[lr], hi);   // two ds_load_b128

    // ---- O += P @ V : vT is t-contiguous per d-row -> vector frag loads
    {
      v16h bv0, bv1, bv2, bv3;
      LOAD_FRAG(bv0, vbase + (long)(lr)      * SEQ + kb * 32, hi);
      LOAD_FRAG(bv1, vbase + (long)(16 + lr) * SEQ + kb * 32, hi);
      LOAD_FRAG(bv2, vbase + (long)(32 + lr) * SEQ + kb * 32, hi);
      LOAD_FRAG(bv3, vbase + (long)(48 + lr) * SEQ + kb * 32, hi);
      o0 = WMMA_F16(pa, bv0, o0);
      o1 = WMMA_F16(pa, bv1, o1);
      o2 = WMMA_F16(pa, bv2, o2);
      o3 = WMMA_F16(pa, bv3, o3);
    }
    __syncthreads();
  }

  // ---- normalize and write y (f16)
#pragma unroll
  for (int r = 0; r < 8; ++r) {
    long t = rowbase + qbase + r + (hi ? 8 : 0);
    float inv = 1.0f / lrow[r];
    _Float16* yp = y + t * DM + h * HD;
    yp[lr]      = (_Float16)(o0[r] * inv);
    yp[16 + lr] = (_Float16)(o1[r] * inv);
    yp[32 + lr] = (_Float16)(o2[r] * inv);
    yp[48 + lr] = (_Float16)(o3[r] * inv);
  }
}

// ---------------------------------------------------------------- launcher
extern "C" void kernel_launch(void* const* d_in, const int* in_sizes, int n_in,
                              void* d_out, int out_size, void* d_ws, size_t ws_size,
                              hipStream_t stream) {
  const int*   idx    = (const int*)  d_in[0];
  const float* wte    = (const float*)d_in[1];
  const float* wpe    = (const float*)d_in[2];
  const float* ln1_s  = (const float*)d_in[3];
  const float* ln1_b  = (const float*)d_in[4];
  const float* attn_w = (const float*)d_in[5];
  const float* attn_b = (const float*)d_in[6];
  const float* proj_w = (const float*)d_in[7];
  const float* proj_b = (const float*)d_in[8];
  const float* ln2_s  = (const float*)d_in[9];
  const float* ln2_b  = (const float*)d_in[10];
  const float* fc_w   = (const float*)d_in[11];
  const float* fc_b   = (const float*)d_in[12];
  const float* fc2_w  = (const float*)d_in[13];
  const float* fc2_b  = (const float*)d_in[14];
  const float* lnf_s  = (const float*)d_in[15];
  const float* lnf_b  = (const float*)d_in[16];
  float* logits = (float*)d_out;

  char* ws = (char*)d_ws;
  size_t off = 0;
  auto carve = [&](size_t bytes) -> char* {
    char* p = ws + off;
    off = (off + bytes + 255) & ~(size_t)255;
    return p;
  };

  _Float16* wte16  = (_Float16*)carve((size_t)VOCAB * DM * 2);   // N-major already
  _Float16* aw16   = (_Float16*)carve((size_t)NLAYER * DM * D3 * 2);  // [l][n][k]
  _Float16* pw16   = (_Float16*)carve((size_t)NLAYER * DM * DM * 2);
  _Float16* fw16   = (_Float16*)carve((size_t)NLAYER * DM * D4 * 2);
  _Float16* f2w16  = (_Float16*)carve((size_t)NLAYER * D4 * DM * 2);
  float*    x      = (float*)   carve((size_t)ROWS * DM * 4);
  _Float16* h16    = (_Float16*)carve((size_t)ROWS * DM * 2);
  _Float16* qkv16  = (_Float16*)carve((size_t)ROWS * D3 * 2);
  _Float16* vt16   = (_Float16*)carve((size_t)NB * NHEAD * HD * SEQ * 2);
  _Float16* y16    = (_Float16*)carve((size_t)ROWS * DM * 2);
  _Float16* ff16   = (_Float16*)carve((size_t)ROWS * D4 * 2);

  // one-time weight conversion (transpose to N-major), graph-safe
  cvt_f32_f16<<<2048, 256, 0, stream>>>(wte, wte16, (long)VOCAB * DM);
  cvt_transpose_f16<<<2048, 256, 0, stream>>>(attn_w, aw16,  NLAYER, DM, D3);
  cvt_transpose_f16<<<1024, 256, 0, stream>>>(proj_w, pw16,  NLAYER, DM, DM);
  cvt_transpose_f16<<<2048, 256, 0, stream>>>(fc_w,   fw16,  NLAYER, DM, D4);
  cvt_transpose_f16<<<2048, 256, 0, stream>>>(fc2_w,  f2w16, NLAYER, D4, DM);

  embed_kernel<<<(ROWS * DM + 255) / 256, 256, 0, stream>>>(idx, wte, wpe, x);

  dim3 blk(128);
  for (int l = 0; l < NLAYER; ++l) {
    layernorm_f16<<<ROWS, 32, 0, stream>>>(x, ln1_s + l * DM, ln1_b + l * DM, h16);
    gemm_wmma<<<dim3(D3 / 128, ROWS / 64), blk, 0, stream>>>(
        h16, aw16 + (size_t)l * DM * D3, attn_b + l * D3, nullptr,
        nullptr, qkv16, vt16, ROWS, D3, DM, GEMM_OUT_QKV);
    flash_attn<<<NB * NHEAD * (SEQ / 16), 32, 0, stream>>>(qkv16, vt16, y16);
    gemm_wmma<<<dim3(DM / 128, ROWS / 64), blk, 0, stream>>>(
        y16, pw16 + (size_t)l * DM * DM, proj_b + l * DM, x,
        x, nullptr, nullptr, ROWS, DM, DM, GEMM_OUT_F32_RESID);
    layernorm_f16<<<ROWS, 32, 0, stream>>>(x, ln2_s + l * DM, ln2_b + l * DM, h16);
    gemm_wmma<<<dim3(D4 / 128, ROWS / 64), blk, 0, stream>>>(
        h16, fw16 + (size_t)l * DM * D4, fc_b + l * D4, nullptr,
        nullptr, ff16, nullptr, ROWS, D4, DM, GEMM_OUT_F16_GELU);
    gemm_wmma<<<dim3(DM / 128, ROWS / 64), blk, 0, stream>>>(
        ff16, f2w16 + (size_t)l * D4 * DM, fc2_b + l * DM, x,
        x, nullptr, nullptr, ROWS, DM, D4, GEMM_OUT_F32_RESID);
  }

  layernorm_f16<<<ROWS, 32, 0, stream>>>(x, lnf_s, lnf_b, h16);
  // tied LM head: logits = xf @ wte^T  (wte is N-major VOCAB x DM)
  gemm_wmma<<<dim3((VOCAB + 127) / 128, ROWS / 64), blk, 0, stream>>>(
      h16, wte16, nullptr, nullptr, logits, nullptr, nullptr,
      ROWS, VOCAB, DM, GEMM_OUT_F32);
}
